// Net_7687991460163
// MI455X (gfx1250) — matrix operand order repaired and checked by reading, not compiled, not run on hardware
//
#include <hip/hip_runtime.h>
#include <math.h>

typedef __attribute__((ext_vector_type(16))) __bf16 v16bf;
typedef __attribute__((ext_vector_type(8)))  float  v8f;

#define BGR   256      // graphs
#define NPGC  200      // nodes per graph at stage 1
#define N0C   (BGR*NPGC)
#define EC    (N0C*16)
#define HC    256

// ---------- weight convert: W[K x N] f32 -> Wt[Npad x K] bf16 (transposed, zero pad cols) ----
__global__ void k_wt(const float* __restrict__ W, __bf16* __restrict__ Wt, int K, int N, int Npad) {
  int i = blockIdx.x * 256 + threadIdx.x;
  if (i >= Npad * K) return;
  int n = i / K, k = i - n * K;
  float v = (n < N) ? W[(size_t)k * N + n] : 0.f;
  Wt[i] = (__bf16)v;
}

// ---------- edge init (copy inputs to mutable workspace, em=1) --------------------------------
__global__ void k_edge_init(const int* __restrict__ es, const int* __restrict__ ed,
                            int* __restrict__ s, int* __restrict__ d, float* __restrict__ em, int E) {
  int e = blockIdx.x * 256 + threadIdx.x; if (e >= E) return;
  s[e] = es[e]; d[e] = ed[e]; em[e] = 1.f;
}

__global__ void k_fill(float* __restrict__ p, float v, int n) {
  int i = blockIdx.x * 256 + threadIdx.x; if (i < n) p[i] = v;
}

__global__ void k_deg(const int* __restrict__ d, const float* __restrict__ em,
                      float* __restrict__ deg, int E) {
  int e = blockIdx.x * 256 + threadIdx.x; if (e >= E) return;
  float m = em[e]; if (m != 0.f) atomicAdd(&deg[d[e]], m);
}

__global__ void k_coef(const int* __restrict__ s, const int* __restrict__ d,
                       const float* __restrict__ em, const float* __restrict__ deg,
                       float* __restrict__ coef, int E) {
  int e = blockIdx.x * 256 + threadIdx.x; if (e >= E) return;
  coef[e] = em[e] * rsqrtf(deg[s[e]]) * rsqrtf(deg[d[e]]);
}

// ---------- WMMA GEMM: C[MxNout] f32 = A[MxK] f32 * Wt[Nout x K] bf16 (+bias)(+relu) ----------
// Each wave computes a 16 x (16*NT) output stripe: one A fragment (loaded f32, converted to
// bf16 in registers) feeds NT v_wmma_f32_16x16x32_bf16 issues -> 4x fewer A loads/converts
// per matrix op. Tile selection is wave-uniform so EXEC stays all-ones for WMMA.
template<int NT>
__global__ void k_gemm(const float* __restrict__ A, const __bf16* __restrict__ Wt,
                       float* __restrict__ C, const float* __restrict__ bias,
                       int M, int K, int Nout, int relu) {
  int wave = threadIdx.x >> 5;
  int lane = threadIdx.x & 31;
  int ntiles  = Nout >> 4;
  int ngroups = ntiles / NT;
  int grp = blockIdx.x * 8 + wave;
  int total = (M >> 4) * ngroups;
  if (grp >= total) return;                  // wave-uniform branch
  int mt = grp / ngroups, ng = grp - mt * ngroups;
  int lo = lane & 15, hi = lane >> 4;

  const float* ap = A + (size_t)(mt * 16 + lo) * K + hi * 16;
  const __bf16* bp[NT];
  #pragma unroll
  for (int j = 0; j < NT; ++j)
    bp[j] = Wt + (size_t)(ng * NT * 16 + j * 16 + lo) * K + hi * 16;

  v8f acc[NT];
  #pragma unroll
  for (int j = 0; j < NT; ++j) acc[j] = (v8f){0.f,0.f,0.f,0.f,0.f,0.f,0.f,0.f};

  for (int kk = 0; kk < K; kk += 32) {
    const float4* a0p = (const float4*)(ap + kk);        // K = kk+hi*16 .. +8
    const float4* a1p = (const float4*)(ap + kk + 8);    // K = kk+hi*16+8 .. +8
    float4 x0 = a0p[0], x1 = a0p[1], x2 = a1p[0], x3 = a1p[1];
    v16bf af;
    af[0]=(__bf16)x0.x; af[1]=(__bf16)x0.y; af[2]=(__bf16)x0.z; af[3]=(__bf16)x0.w;
    af[4]=(__bf16)x1.x; af[5]=(__bf16)x1.y; af[6]=(__bf16)x1.z; af[7]=(__bf16)x1.w;
    af[8]=(__bf16)x2.x; af[9]=(__bf16)x2.y; af[10]=(__bf16)x2.z; af[11]=(__bf16)x2.w;
    af[12]=(__bf16)x3.x; af[13]=(__bf16)x3.y; af[14]=(__bf16)x3.z; af[15]=(__bf16)x3.w;
    #pragma unroll
    for (int j = 0; j < NT; ++j) {
      v16bf bf = *(const v16bf*)(bp[j] + kk);            // 16 contiguous bf16 (32B aligned)
      acc[j] = __builtin_amdgcn_wmma_f32_16x16x32_bf16(false, af, false, bf,
                                                       (short)0, acc[j], false, false);
    }
  }

  #pragma unroll
  for (int j = 0; j < NT; ++j) {
    int bcol = ng * NT * 16 + j * 16 + lo;
    float bb = bias ? bias[bcol] : 0.f;
    #pragma unroll
    for (int r = 0; r < 8; ++r) {
      int row = mt * 16 + r + hi * 8;
      float v = acc[j][r] + bb;
      if (relu) v = fmaxf(v, 0.f);
      C[(size_t)row * Nout + bcol] = v;
    }
  }
}

// ---------- GCN finalize: h = xw/deg + b, then edge atomics, then relu ------------------------
__global__ void k_gcn_init(const float* __restrict__ xw, const float* __restrict__ deg,
                           const float* __restrict__ b, float* __restrict__ h) {
  int n = blockIdx.x, f = threadIdx.x;
  size_t i = (size_t)n * HC + f;
  h[i] = xw[i] * (1.f / deg[n]) + b[f];
}

__global__ void k_gcn_edge(const int* __restrict__ s, const int* __restrict__ d,
                           const float* __restrict__ coef, const float* __restrict__ xw,
                           float* __restrict__ h) {
  int e = blockIdx.x;
  float c = coef[e];
  if (c == 0.f) return;
  int src = s[e], dst = d[e], f = threadIdx.x;
  atomicAdd(&h[(size_t)dst * HC + f], xw[(size_t)src * HC + f] * c);
}

__global__ void k_relu(float* __restrict__ p, long long n) {
  long long i = (long long)blockIdx.x * 256 + threadIdx.x;
  if (i < n) p[i] = fmaxf(p[i], 0.f);
}

// ---------- score GCN (H -> 1): wave-per-node dot, then same deg/coef pipeline ----------------
__global__ void k_score_xw(const float* __restrict__ h, const float* __restrict__ Ws,
                           float* __restrict__ sxw, int N) {
  int wave = threadIdx.x >> 5, lane = threadIdx.x & 31;
  int n = blockIdx.x * 8 + wave; if (n >= N) return;
  const float* r = h + (size_t)n * HC;
  float sum = 0.f;
  #pragma unroll
  for (int j = 0; j < 8; ++j) sum += r[lane + 32 * j] * Ws[lane + 32 * j];
  for (int m = 16; m; m >>= 1) sum += __shfl_xor(sum, m, 32);
  if (lane == 0) sxw[n] = sum;
}

__global__ void k_score_init(const float* __restrict__ sxw, const float* __restrict__ deg,
                             const float* __restrict__ bs, float* __restrict__ sc, int N) {
  int n = blockIdx.x * 256 + threadIdx.x; if (n >= N) return;
  sc[n] = sxw[n] * (1.f / deg[n]) + bs[0];
}

__global__ void k_score_edge(const int* __restrict__ s, const int* __restrict__ d,
                             const float* __restrict__ coef, const float* __restrict__ sxw,
                             float* __restrict__ sc, int E) {
  int e = blockIdx.x * 256 + threadIdx.x; if (e >= E) return;
  float c = coef[e]; if (c != 0.f) atomicAdd(&sc[d[e]], sxw[s[e]] * c);
}

// ---------- top-k per graph by rank (order-invariant downstream) ------------------------------
__global__ void k_pool_select(const float* __restrict__ sc, int* __restrict__ remap,
                              int* __restrict__ sel, float* __restrict__ gate, int npg, int k) {
  __shared__ float ss[256];
  int g = blockIdx.x, t = threadIdx.x;
  if (t < npg) ss[t] = sc[g * npg + t];
  __syncthreads();
  if (t >= npg) return;
  float si = ss[t];
  int rank = 0;
  for (int j = 0; j < npg; ++j) {
    float sj = ss[j];
    rank += (sj > si) || (sj == si && j < t);
  }
  int node = g * npg + t;
  if (rank < k) {
    int p = g * k + rank;
    remap[node] = p; sel[p] = node; gate[p] = tanhf(si);
  } else remap[node] = -1;
}

__global__ void k_pool_gather(const float* __restrict__ h, const int* __restrict__ sel,
                              const float* __restrict__ gate, float* __restrict__ xn) {
  int p = blockIdx.x, f = threadIdx.x;
  xn[(size_t)p * HC + f] = h[(size_t)sel[p] * HC + f] * gate[p];
}

__global__ void k_readout(const float* __restrict__ xn, float* __restrict__ z, int k) {
  int g = blockIdx.x, f = threadIdx.x;
  const float* base = xn + (size_t)g * k * HC + f;
  float mx = -INFINITY, sm = 0.f;
  for (int i = 0; i < k; ++i) { float v = base[(size_t)i * HC]; mx = fmaxf(mx, v); sm += v; }
  z[(size_t)g * 512 + f]       += mx;
  z[(size_t)g * 512 + 256 + f] += sm / (float)k;
}

__global__ void k_edge_remap(int* __restrict__ s, int* __restrict__ d, float* __restrict__ em,
                             const int* __restrict__ remap, int E) {
  int e = blockIdx.x * 256 + threadIdx.x; if (e >= E) return;
  int ns = remap[s[e]], nd = remap[d[e]];
  float v = em[e] * ((ns >= 0 && nd >= 0) ? 1.f : 0.f);
  s[e] = ns < 0 ? 0 : ns;
  d[e] = nd < 0 ? 0 : nd;
  em[e] = v;
}

__global__ void k_logsoftmax(const float* __restrict__ t3, const float* __restrict__ bl3,
                             float* __restrict__ out) {
  int g = threadIdx.x;   // 256 graphs, one block of 256
  float v[10]; float mx = -1e30f;
  #pragma unroll
  for (int c = 0; c < 10; ++c) { v[c] = t3[g * 16 + c] + bl3[c]; mx = fmaxf(mx, v[c]); }
  float s = 0.f;
  #pragma unroll
  for (int c = 0; c < 10; ++c) s += expf(v[c] - mx);
  float l = logf(s);
  #pragma unroll
  for (int c = 0; c < 10; ++c) out[g * 10 + c] = v[c] - mx - l;
}

// ======================== host launcher =======================================================
extern "C" void kernel_launch(void* const* d_in, const int* in_sizes, int n_in,
                              void* d_out, int out_size, void* d_ws, size_t ws_size,
                              hipStream_t stream) {
  (void)in_sizes; (void)n_in; (void)out_size; (void)ws_size;
  const float* x   = (const float*)d_in[0];
  const int*   es  = (const int*)d_in[1];
  const int*   ed  = (const int*)d_in[2];
  const float* W1  = (const float*)d_in[3];  const float* b1  = (const float*)d_in[4];
  const float* Ws1 = (const float*)d_in[5];  const float* bs1 = (const float*)d_in[6];
  const float* W2  = (const float*)d_in[7];  const float* b2  = (const float*)d_in[8];
  const float* Ws2 = (const float*)d_in[9];  const float* bs2 = (const float*)d_in[10];
  const float* W3  = (const float*)d_in[11]; const float* b3  = (const float*)d_in[12];
  const float* Ws3 = (const float*)d_in[13]; const float* bs3 = (const float*)d_in[14];
  const float* Wl1 = (const float*)d_in[15]; const float* bl1 = (const float*)d_in[16];
  const float* Wl2 = (const float*)d_in[17]; const float* bl2 = (const float*)d_in[18];
  const float* Wl3 = (const float*)d_in[19]; const float* bl3 = (const float*)d_in[20];
  float* out = (float*)d_out;

  const int N0 = N0C, E = EC, N1 = BGR * 100;
  const int EB = E / 256;

  // workspace arena
  char* base = (char*)d_ws; size_t off = 0;
  auto alloc = [&](size_t bytes) -> void* {
    off = (off + 255) & ~(size_t)255;
    void* r = base + off; off += bytes; return r;
  };
  float*  bufA  = (float*)alloc((size_t)N0 * HC * 4);
  float*  bufB  = (float*)alloc((size_t)N0 * HC * 4);
  int*    s_ws  = (int*)  alloc((size_t)E * 4);
  int*    d_wsv = (int*)  alloc((size_t)E * 4);
  float*  em_ws = (float*)alloc((size_t)E * 4);
  float*  coef  = (float*)alloc((size_t)E * 4);
  float*  deg   = (float*)alloc((size_t)N0 * 4);
  float*  score = (float*)alloc((size_t)N0 * 4);
  float*  sxw   = (float*)alloc((size_t)N0 * 4);
  int*    remap = (int*)  alloc((size_t)N0 * 4);
  int*    sel   = (int*)  alloc((size_t)N1 * 4);
  float*  gate  = (float*)alloc((size_t)N1 * 4);
  float*  z     = (float*)alloc((size_t)BGR * 512 * 4);
  float*  t1    = (float*)alloc((size_t)BGR * 256 * 4);
  float*  t2    = (float*)alloc((size_t)BGR * 128 * 4);
  float*  t3    = (float*)alloc((size_t)BGR * 16 * 4);
  __bf16* Wt1   = (__bf16*)alloc((size_t)256 * 128 * 2);
  __bf16* Wt2   = (__bf16*)alloc((size_t)256 * 256 * 2);
  __bf16* Wt3   = (__bf16*)alloc((size_t)256 * 256 * 2);
  __bf16* Wtl1  = (__bf16*)alloc((size_t)256 * 512 * 2);
  __bf16* Wtl2  = (__bf16*)alloc((size_t)128 * 256 * 2);
  __bf16* Wtl3  = (__bf16*)alloc((size_t)16  * 128 * 2);

  // weights -> transposed bf16 (once per launch, deterministic)
  k_wt<<<(256*128+255)/256, 256, 0, stream>>>(W1,  Wt1,  128, 256, 256);
  k_wt<<<(256*256+255)/256, 256, 0, stream>>>(W2,  Wt2,  256, 256, 256);
  k_wt<<<(256*256+255)/256, 256, 0, stream>>>(W3,  Wt3,  256, 256, 256);
  k_wt<<<(256*512+255)/256, 256, 0, stream>>>(Wl1, Wtl1, 512, 256, 256);
  k_wt<<<(128*256+255)/256, 256, 0, stream>>>(Wl2, Wtl2, 256, 128, 128);
  k_wt<<<(16*128+255)/256,  256, 0, stream>>>(Wl3, Wtl3, 128, 10,  16);

  k_edge_init<<<EB, 256, 0, stream>>>(es, ed, s_ws, d_wsv, em_ws, E);
  k_fill<<<(BGR*512+255)/256, 256, 0, stream>>>(z, 0.f, BGR * 512);

  auto gemm = [&](const float* A, const __bf16* Wt, float* C, const float* bias,
                  int M, int K, int Nout, int relu) {
    int ntiles = Nout / 16;
    if (ntiles % 4 == 0) {
      int groups = (M / 16) * (ntiles / 4);
      k_gemm<4><<<(groups + 7) / 8, 256, 0, stream>>>(A, Wt, C, bias, M, K, Nout, relu);
    } else if (ntiles % 2 == 0) {
      int groups = (M / 16) * (ntiles / 2);
      k_gemm<2><<<(groups + 7) / 8, 256, 0, stream>>>(A, Wt, C, bias, M, K, Nout, relu);
    } else {
      int groups = (M / 16) * ntiles;
      k_gemm<1><<<(groups + 7) / 8, 256, 0, stream>>>(A, Wt, C, bias, M, K, Nout, relu);
    }
  };

  auto stage = [&](const float* Ain, int N, int Kin, const __bf16* Wt, const float* b,
                   const float* Ws, const float* bs, int npg, int kkeep,
                   float* xwbuf, float* hbuf, float* xnew) {
    k_fill<<<(N+255)/256, 256, 0, stream>>>(deg, 1.f, N);
    k_deg<<<EB, 256, 0, stream>>>(d_wsv, em_ws, deg, E);
    k_coef<<<EB, 256, 0, stream>>>(s_ws, d_wsv, em_ws, deg, coef, E);
    gemm(Ain, Wt, xwbuf, nullptr, N, Kin, HC, 0);
    k_gcn_init<<<N, 256, 0, stream>>>(xwbuf, deg, b, hbuf);
    k_gcn_edge<<<E, 256, 0, stream>>>(s_ws, d_wsv, coef, xwbuf, hbuf);
    k_relu<<<(unsigned)(((long long)N*HC + 255)/256), 256, 0, stream>>>(hbuf, (long long)N*HC);
    k_score_xw<<<(N+7)/8, 256, 0, stream>>>(hbuf, Ws, sxw, N);
    k_score_init<<<(N+255)/256, 256, 0, stream>>>(sxw, deg, bs, score, N);
    k_score_edge<<<EB, 256, 0, stream>>>(s_ws, d_wsv, coef, sxw, score, E);
    k_pool_select<<<BGR, 256, 0, stream>>>(score, remap, sel, gate, npg, kkeep);
    k_pool_gather<<<BGR * kkeep, 256, 0, stream>>>(hbuf, sel, gate, xnew);
    k_readout<<<BGR, 256, 0, stream>>>(xnew, z, kkeep);
    k_edge_remap<<<EB, 256, 0, stream>>>(s_ws, d_wsv, em_ws, remap, E);
  };

  // stage 1: A=x (f32), xw->bufA, h->bufB, x_new->bufA
  stage(x,    N0, 128, Wt1, b1, Ws1, bs1, NPGC, 100, bufA, bufB, bufA);
  // stage 2: A=bufA, xw->bufB, h->bufA, x_new->bufB
  stage(bufA, N1, 256, Wt2, b2, Ws2, bs2, 100,  50,  bufB, bufA, bufB);
  // stage 3: A=bufB, xw->bufA, h->bufB, x_new->bufA
  stage(bufB, BGR*50, 256, Wt3, b3, Ws3, bs3, 50, 25, bufA, bufB, bufA);

  // MLP head (all WMMA)
  gemm(z,  Wtl1, t1, bl1,     BGR, 512, 256, 1);
  gemm(t1, Wtl2, t2, bl2,     BGR, 256, 128, 1);
  gemm(t2, Wtl3, t3, nullptr, BGR, 128, 16,  0);
  k_logsoftmax<<<1, 256, 0, stream>>>(t3, bl3, out);
}
